// STNDTEncoder_86973087744568
// MI455X (gfx1250) — matrix operand order repaired and checked by reading, not compiled, not run on hardware
//
#include <hip/hip_runtime.h>

// ---------------------------------------------------------------------------
// STNDT encoder block for gfx1250 (MI455X), bf16 WMMA with f32 accumulation.
// B=4, T=1024, C=1024, H=16, DH=DS=64, F=4096.
// This revision: double-buffered GLOBAL_LOAD_ASYNC_TO_LDS_B128 tile staging
// (ASYNCcnt) in the GEMM + K-tile staging of both attention kernels.
// ---------------------------------------------------------------------------

static constexpr int BATCH = 4;
static constexpr int SEQ   = 1024;   // T
static constexpr int CH    = 1024;   // C
static constexpr int NH    = 16;     // heads
static constexpr int HD    = 64;     // head dim (temporal DH == spatial DS)
static constexpr int FF    = 4096;   // FFN hidden

typedef __attribute__((ext_vector_type(16))) __bf16 v16bf;
typedef __attribute__((ext_vector_type(8)))  float  v8f;

union Frag16 {
    v16bf bf;
    uint4 q[2];
    unsigned short h[16];
};

__device__ __forceinline__ unsigned short f2bf(float f) {
    unsigned u = __float_as_uint(f);
    unsigned r = (u + 0x7FFFu + ((u >> 16) & 1u)) >> 16;   // round-nearest-even
    return (unsigned short)r;
}

// ---- CDNA5 async global->LDS copy (tracked by ASYNCcnt) --------------------
// lds_off: byte offset in LDS (low 32 bits of the generic pointer to shared);
// gaddr  : 64-bit global address. 16 bytes per active lane.
__device__ __forceinline__ void async_copy_b128(unsigned lds_off, const void* gaddr) {
    unsigned long long ga = (unsigned long long)(size_t)gaddr;
    asm volatile("global_load_async_to_lds_b128 %0, %1, off"
                 :: "v"(lds_off), "v"(ga) : "memory");
}
__device__ __forceinline__ void wait_async0() {
    asm volatile("s_wait_asynccnt 0x0" ::: "memory");
}
__device__ __forceinline__ unsigned lds_off_of(const void* p) {
    return (unsigned)(size_t)p;    // flat LDS address: offset lives in addr[31:0]
}

// ---------------------------------------------------------------------------
// f32 -> bf16 elementwise convert (weights are (out,in) row-major == the (N,K)
// layout the GEMM B-tile staging wants, so no transpose is ever needed).
// ---------------------------------------------------------------------------
__global__ void k_cvt(const float* __restrict__ in, unsigned short* __restrict__ out, long n) {
    long i = (long)blockIdx.x * blockDim.x + threadIdx.x;
    long stride = (long)gridDim.x * blockDim.x;
    for (; i < n; i += stride) out[i] = f2bf(in[i]);
}

// ---------------------------------------------------------------------------
// Row LayerNorm (width = 1024 in all uses), writes bf16 (feeds GEMMs only).
// ---------------------------------------------------------------------------
__global__ __launch_bounds__(256)
void k_layernorm(const float* __restrict__ x, const float* __restrict__ g,
                 const float* __restrict__ bta, unsigned short* __restrict__ out, int width) {
    __shared__ float sred[20];
    int row = blockIdx.x;
    const float* xr = x + (size_t)row * width;
    float lsum = 0.f, lsq = 0.f;
    for (int i = threadIdx.x; i < width; i += 256) { float v = xr[i]; lsum += v; lsq += v * v; }
    for (int m = 16; m >= 1; m >>= 1) { lsum += __shfl_xor(lsum, m, 32); lsq += __shfl_xor(lsq, m, 32); }
    int wave = threadIdx.x >> 5;
    if ((threadIdx.x & 31) == 0) { sred[wave] = lsum; sred[wave + 8] = lsq; }
    __syncthreads();
    if (threadIdx.x == 0) {
        float a = 0.f, b = 0.f;
        for (int w = 0; w < 8; ++w) { a += sred[w]; b += sred[w + 8]; }
        sred[16] = a; sred[17] = b;
    }
    __syncthreads();
    float mean = sred[16] / width;
    float var  = sred[17] / width - mean * mean;
    float inv  = rsqrtf(var + 1e-6f);
    for (int i = threadIdx.x; i < width; i += 256) {
        float v = (xr[i] - mean) * inv * g[i] + bta[i];
        out[(size_t)row * width + i] = f2bf(v);
    }
}

// ---------------------------------------------------------------------------
// Tiled bf16 WMMA GEMM:  out(M,N) = A(M,K) @ Bw(N,K)^T  [Bw = (out,in) weight]
// Optional epilogue: +bias[col], PReLU(alpha), +residual(M,N f32),
// write f32 and/or bf16. 256 threads = 8 waves, tile 128x128x32,
// each wave owns a 64x32 sub-tile = 4x2 WMMA accumulators.
// Double-buffered async global->LDS staging: tile k+1 streams in via the
// async DMA path while tile k is consumed by the WMMA pipe.
// ---------------------------------------------------------------------------
__global__ __launch_bounds__(256)
void k_gemm(const unsigned short* __restrict__ A, const unsigned short* __restrict__ Bw,
            const float* __restrict__ bias, const float* __restrict__ alphap,
            const float* __restrict__ residual, float* __restrict__ outF,
            unsigned short* __restrict__ outB, int M, int N, int K) {
    constexpr int LDA = 40;                   // 80B row stride: 16B-aligned, bank-skewed
    __shared__ unsigned short sA[2][128 * LDA];
    __shared__ unsigned short sB[2][128 * LDA];

    int tid = threadIdx.x;
    int wave = tid >> 5, lane = tid & 31, lh = lane & 15;
    int waveM = wave >> 2, waveN = wave & 3;
    int rowBlock = blockIdx.y * 128;
    int colBlock = blockIdx.x * 128;
    int kA0  = (lane < 16) ? 0 : 8;           // A-fragment K origin (ISA 16-bit A 16x32 layout)
    int hsel = (lane < 16) ? 0 : 8;           // B-fragment K origin (same K order as A fragment)

    // per-thread staging slots: 2 x 16B chunks of A and of B per tile
    const unsigned short* gA[2];
    const unsigned short* gB[2];
    unsigned dA[2][2], dB[2][2];              // [buffer][chunk]
    for (int i = 0; i < 2; ++i) {
        int idx = tid + i * 256;              // 0..511
        int r = idx >> 2, seg = idx & 3;      // row, 8-half column segment
        gA[i] = A  + (size_t)(rowBlock + r) * K + seg * 8;
        gB[i] = Bw + (size_t)(colBlock + r) * K + seg * 8;
        for (int bsel = 0; bsel < 2; ++bsel) {
            dA[bsel][i] = lds_off_of(&sA[bsel][r * LDA + seg * 8]);
            dB[bsel][i] = lds_off_of(&sB[bsel][r * LDA + seg * 8]);
        }
    }
    auto issue_tile = [&](int bsel, int k0) {
        for (int i = 0; i < 2; ++i) {
            async_copy_b128(dA[bsel][i], gA[i] + k0);
            async_copy_b128(dB[bsel][i], gB[i] + k0);
        }
    };

    v8f acc[4][2] = {};
    int nk = K >> 5;
    issue_tile(0, 0);

    for (int kt = 0; kt < nk; ++kt) {
        wait_async0();                        // this wave's DMA into buf kt&1 done
        __syncthreads();                      // everyone's DMA visible
        if (kt + 1 < nk) issue_tile((kt + 1) & 1, (kt + 1) * 32);

        const unsigned short* cA = sA[kt & 1];
        const unsigned short* cB = sB[kt & 1];
        Frag16 bfr[2];
        for (int ni = 0; ni < 2; ++ni) {
            int n = waveN * 32 + ni * 16 + lh;
            bfr[ni].q[0] = *reinterpret_cast<const uint4*>(&cB[n * LDA + hsel]);
            bfr[ni].q[1] = *reinterpret_cast<const uint4*>(&cB[n * LDA + hsel + 16]);
        }
        for (int mi = 0; mi < 4; ++mi) {
            int m = waveM * 64 + mi * 16 + lh;
            Frag16 a;
            a.q[0] = *reinterpret_cast<const uint4*>(&cA[m * LDA + kA0]);
            a.q[1] = *reinterpret_cast<const uint4*>(&cA[m * LDA + kA0 + 16]);
            for (int ni = 0; ni < 2; ++ni)
                acc[mi][ni] = __builtin_amdgcn_wmma_f32_16x16x32_bf16(
                    false, a.bf, false, bfr[ni].bf, (short)0, acc[mi][ni], false, false);
        }
        __syncthreads();                      // reads of buf kt&1 done before its re-fill
    }

    // epilogue (C/D layout: lanes 0-15 rows r, lanes 16-31 rows r+8, col = lh)
    float alpha = alphap ? *alphap : 0.f;
    int mloc0 = waveM * 64 + ((lane < 16) ? 0 : 8);
    for (int mi = 0; mi < 4; ++mi) {
        for (int ni = 0; ni < 2; ++ni) {
            int col = colBlock + waveN * 32 + ni * 16 + lh;
            float bval = bias ? bias[col] : 0.f;
            for (int r = 0; r < 8; ++r) {
                int row = rowBlock + mloc0 + mi * 16 + r;
                float c = acc[mi][ni][r] + bval;
                if (alphap) c = (c >= 0.f) ? c : alpha * c;
                if (residual) c += residual[(size_t)row * N + col];
                if (outF) outF[(size_t)row * N + col] = c;
                if (outB) outB[(size_t)row * N + col] = f2bf(c);
            }
        }
    }
}

// ---------------------------------------------------------------------------
// Causal flash attention (temporal branch). One block = 128 threads (4 waves)
// handles 64 query rows of one (b,h). Online softmax; P re-fragmented via
// per-wave LDS staging (C-layout -> A-layout). K tile streamed via async DMA;
// V tile transposed through registers. Writes o as bf16 into (B*T,C).
// ---------------------------------------------------------------------------
__global__ __launch_bounds__(128)
void k_flash(const unsigned short* __restrict__ Q, const unsigned short* __restrict__ Kb,
             const unsigned short* __restrict__ Vb, unsigned short* __restrict__ O) {
    constexpr int LKV = 72;                       // 144B stride, 16B aligned
    __shared__ unsigned short sK [64 * LKV];      // (key, d)
    __shared__ unsigned short sVT[64 * LKV];      // (d, key)  -> B operand for P@V
    __shared__ unsigned short sP [4][16 * LKV];   // per-wave P staging

    int tid = threadIdx.x, wave = tid >> 5, lane = tid & 31, lh = lane & 15;
    int qb = blockIdx.x & 15;                     // SEQ/64
    int bh = blockIdx.x >> 4;
    int b = bh >> 4, h = bh & 15;
    int qrow0 = qb * 64 + wave * 16;
    int headoff = h * HD;
    int kA0  = (lane < 16) ? 0 : 8;
    int hsel = (lane < 16) ? 0 : 8;
    int mhalf = (lane < 16) ? 0 : 8;

    // per-thread K staging slots (fixed LDS dest, varying global src)
    int stKey[4], stSeg[4];
    unsigned dK[4];
    for (int i = 0; i < 4; ++i) {
        int idx = tid + i * 128;
        stKey[i] = idx >> 3; stSeg[i] = idx & 7;
        dK[i] = lds_off_of(&sK[stKey[i] * LKV + stSeg[i] * 8]);
    }

    Frag16 aq[2];
    {
        const unsigned short* qrow = Q + ((size_t)(b * SEQ + qrow0 + lh)) * CH + headoff;
        for (int kk = 0; kk < 2; ++kk) {
            aq[kk].q[0] = *reinterpret_cast<const uint4*>(qrow + kk * 32 + kA0);
            aq[kk].q[1] = *reinterpret_cast<const uint4*>(qrow + kk * 32 + kA0 + 16);
        }
    }

    v8f oacc[4] = {};
    float rm[8], rl[8];
    for (int r = 0; r < 8; ++r) { rm[r] = -3.0e38f; rl[r] = 0.f; }

    for (int kb = 0; kb <= qb; ++kb) {
        // stage K tile (async DMA) and transposed V tile (register path)
        for (int i = 0; i < 4; ++i) {
            const size_t rowoff = ((size_t)(b * SEQ + kb * 64 + stKey[i])) * CH + headoff + stSeg[i] * 8;
            async_copy_b128(dK[i], Kb + rowoff);
            uint4 vv = *reinterpret_cast<const uint4*>(Vb + rowoff);
            const unsigned short* vh = reinterpret_cast<const unsigned short*>(&vv);
            for (int j = 0; j < 8; ++j) sVT[(stSeg[i] * 8 + j) * LKV + stKey[i]] = vh[j];
        }
        wait_async0();
        __syncthreads();

        // scores: 16 rows x 64 keys, K = HD = 64 (2 WMMA k-steps x 4 key tiles)
        v8f sacc[4] = {};
        for (int kk = 0; kk < 2; ++kk) {
            for (int nt = 0; nt < 4; ++nt) {
                Frag16 bk;
                int n = nt * 16 + lh;
                bk.q[0] = *reinterpret_cast<const uint4*>(&sK[n * LKV + kk * 32 + hsel]);
                bk.q[1] = *reinterpret_cast<const uint4*>(&sK[n * LKV + kk * 32 + hsel + 16]);
                sacc[nt] = __builtin_amdgcn_wmma_f32_16x16x32_bf16(
                    false, aq[kk].bf, false, bk.bf, (short)0, sacc[nt], false, false);
            }
        }
        // scale + causal mask
        int mabs = qb * 64 + wave * 16 + mhalf;
        for (int nt = 0; nt < 4; ++nt) {
            int key = kb * 64 + nt * 16 + lh;
            for (int r = 0; r < 8; ++r) {
                float s = sacc[nt][r] * 0.125f;               // DH^-0.5
                if (key > mabs + r) s = -1.0e9f;
                sacc[nt][r] = s;
            }
        }
        // online softmax update (row reductions over 16-lane halves)
        for (int r = 0; r < 8; ++r) {
            float mx = fmaxf(fmaxf(sacc[0][r], sacc[1][r]), fmaxf(sacc[2][r], sacc[3][r]));
            for (int m = 8; m >= 1; m >>= 1) mx = fmaxf(mx, __shfl_xor(mx, m, 32));
            float nm = fmaxf(rm[r], mx);
            float corr = __expf(rm[r] - nm);
            rm[r] = nm;
            float ps = 0.f;
            for (int nt = 0; nt < 4; ++nt) { float p = __expf(sacc[nt][r] - nm); sacc[nt][r] = p; ps += p; }
            for (int m = 8; m >= 1; m >>= 1) ps += __shfl_xor(ps, m, 32);
            rl[r] = rl[r] * corr + ps;
            for (int nt = 0; nt < 4; ++nt) oacc[nt][r] *= corr;
        }
        // re-fragment P through per-wave LDS (wave-private region, same-wave LDS ordering)
        for (int nt = 0; nt < 4; ++nt) {
            int cc = nt * 16 + lh;
            for (int r = 0; r < 8; ++r)
                sP[wave][(mhalf + r) * LKV + cc] = f2bf(sacc[nt][r]);
        }
        // P @ V
        for (int kk = 0; kk < 2; ++kk) {
            Frag16 ap;
            ap.q[0] = *reinterpret_cast<const uint4*>(&sP[wave][lh * LKV + kk * 32 + kA0]);
            ap.q[1] = *reinterpret_cast<const uint4*>(&sP[wave][lh * LKV + kk * 32 + kA0 + 16]);
            for (int nt = 0; nt < 4; ++nt) {
                Frag16 bv;
                int n = nt * 16 + lh;                          // dh column
                bv.q[0] = *reinterpret_cast<const uint4*>(&sVT[n * LKV + kk * 32 + hsel]);
                bv.q[1] = *reinterpret_cast<const uint4*>(&sVT[n * LKV + kk * 32 + hsel + 16]);
                oacc[nt] = __builtin_amdgcn_wmma_f32_16x16x32_bf16(
                    false, ap.bf, false, bv.bf, (short)0, oacc[nt], false, false);
            }
        }
        __syncthreads();
    }

    for (int nt = 0; nt < 4; ++nt)
        for (int r = 0; r < 8; ++r) {
            float o = oacc[nt][r] / rl[r];
            int row = b * SEQ + qb * 64 + wave * 16 + mhalf + r;
            int col = headoff + nt * 16 + lh;
            O[(size_t)row * CH + col] = f2bf(o);
        }
}

// ---------------------------------------------------------------------------
// Spatial attention: SW(b,n,m) = mean_h softmax_m( qs_h @ ks_h^T * DS^-0.5 ).
// One block = 128 threads (4 waves) owns 32 rows (n) of one batch; loops heads.
// Two-pass streaming softmax (pass A: exact row max/sum; pass B: recompute and
// emit normalized probs / H into global SW). Wave w owns key-tile column w.
// K tiles streamed via async DMA.
// ---------------------------------------------------------------------------
__global__ __launch_bounds__(128)
void k_spatial(const unsigned short* __restrict__ QS, const unsigned short* __restrict__ KS,
               float* __restrict__ SW) {
    constexpr int LKV = 72;
    __shared__ unsigned short sK[64 * LKV];
    __shared__ float redM[4][32], redL[4][32], Mst[32], Lst[32];

    int tid = threadIdx.x, wave = tid >> 5, lane = tid & 31, lh = lane & 15;
    int b = blockIdx.x / (CH / 32);
    int rowbase = (blockIdx.x % (CH / 32)) * 32;
    int kA0  = (lane < 16) ? 0 : 8;
    int hsel = (lane < 16) ? 0 : 8;
    int mhalf = (lane < 16) ? 0 : 8;

    int stKey[4], stSeg[4];
    unsigned dK[4];
    for (int i = 0; i < 4; ++i) {
        int idx = tid + i * 128;
        stKey[i] = idx >> 3; stSeg[i] = idx & 7;
        dK[i] = lds_off_of(&sK[stKey[i] * LKV + stSeg[i] * 8]);
    }

    for (int h = 0; h < NH; ++h) {
        Frag16 aq[2][2];
        for (int mt = 0; mt < 2; ++mt) {
            const unsigned short* qrow = QS + ((size_t)(b * CH + rowbase + mt * 16 + lh)) * CH + h * HD;
            for (int kk = 0; kk < 2; ++kk) {
                aq[mt][kk].q[0] = *reinterpret_cast<const uint4*>(qrow + kk * 32 + kA0);
                aq[mt][kk].q[1] = *reinterpret_cast<const uint4*>(qrow + kk * 32 + kA0 + 16);
            }
        }
        float pm[2][8], pl[2][8];
        for (int mt = 0; mt < 2; ++mt)
            for (int r = 0; r < 8; ++r) { pm[mt][r] = -3.0e38f; pl[mt][r] = 0.f; }

        for (int pass = 0; pass < 2; ++pass) {
            for (int mb = 0; mb < CH / 64; ++mb) {
                for (int i = 0; i < 4; ++i)
                    async_copy_b128(dK[i],
                        KS + ((size_t)(b * CH + mb * 64 + stKey[i])) * CH + h * HD + stSeg[i] * 8);
                wait_async0();
                __syncthreads();

                v8f sacc[2] = {};
                for (int kk = 0; kk < 2; ++kk) {
                    Frag16 bk;
                    int n = wave * 16 + lh;                    // this wave's key tile
                    bk.q[0] = *reinterpret_cast<const uint4*>(&sK[n * LKV + kk * 32 + hsel]);
                    bk.q[1] = *reinterpret_cast<const uint4*>(&sK[n * LKV + kk * 32 + hsel + 16]);
                    for (int mt = 0; mt < 2; ++mt)
                        sacc[mt] = __builtin_amdgcn_wmma_f32_16x16x32_bf16(
                            false, aq[mt][kk].bf, false, bk.bf, (short)0, sacc[mt], false, false);
                }

                if (pass == 0) {
                    for (int mt = 0; mt < 2; ++mt)
                        for (int r = 0; r < 8; ++r) {
                            float s = sacc[mt][r] * 0.125f;    // DS^-0.5
                            float mx = s;
                            for (int m = 8; m >= 1; m >>= 1) mx = fmaxf(mx, __shfl_xor(mx, m, 32));
                            float nm = fmaxf(pm[mt][r], mx);
                            float p = __expf(s - nm);
                            float ps = p;
                            for (int m = 8; m >= 1; m >>= 1) ps += __shfl_xor(ps, m, 32);
                            pl[mt][r] = pl[mt][r] * __expf(pm[mt][r] - nm) + ps;
                            pm[mt][r] = nm;
                        }
                } else {
                    for (int mt = 0; mt < 2; ++mt)
                        for (int r = 0; r < 8; ++r) {
                            int row = mt * 16 + mhalf + r;
                            float p = __expf(sacc[mt][r] * 0.125f - Mst[row]) / Lst[row] * (1.0f / NH);
                            size_t gi = ((size_t)(b * CH + rowbase + row)) * CH + mb * 64 + wave * 16 + lh;
                            if (h == 0) SW[gi] = p; else SW[gi] += p;
                        }
                }
                __syncthreads();
            }
            if (pass == 0) {
                if (lane == 0 || lane == 16)
                    for (int mt = 0; mt < 2; ++mt)
                        for (int r = 0; r < 8; ++r) {
                            int row = mt * 16 + mhalf + r;
                            redM[wave][row] = pm[mt][r];
                            redL[wave][row] = pl[mt][r];
                        }
                __syncthreads();
                if (tid < 32) {
                    float m = redM[0][tid];
                    for (int w = 1; w < 4; ++w) m = fmaxf(m, redM[w][tid]);
                    float l = 0.f;
                    for (int w = 0; w < 4; ++w) l += redL[w][tid] * __expf(redM[w][tid] - m);
                    Mst[tid] = m; Lst[tid] = l;
                }
                __syncthreads();
            }
        }
    }
}

// ---------------------------------------------------------------------------
// Orchestration
// ---------------------------------------------------------------------------
extern "C" void kernel_launch(void* const* d_in, const int* in_sizes, int n_in,
                              void* d_out, int out_size, void* d_ws, size_t ws_size,
                              hipStream_t stream) {
    (void)in_sizes; (void)n_in; (void)out_size; (void)ws_size;

    const float* x_T  = (const float*)d_in[0];
    const float* x_S  = (const float*)d_in[1];
    // d_in[2] src_mask: causal mask is applied analytically in k_flash
    const float* Wq_t = (const float*)d_in[3];
    const float* Wk_t = (const float*)d_in[4];
    const float* Wv_t = (const float*)d_in[5];
    const float* Wo   = (const float*)d_in[6];
    const float* Wq_s = (const float*)d_in[7];
    const float* Wk_s = (const float*)d_in[8];
    const float* ln_t1_w = (const float*)d_in[9],  *ln_t1_b = (const float*)d_in[10];
    const float* ln_t2_w = (const float*)d_in[11], *ln_t2_b = (const float*)d_in[12];
    const float* ln_s_w  = (const float*)d_in[13], *ln_s_b  = (const float*)d_in[14];
    const float* ln_f_w  = (const float*)d_in[15], *ln_f_b  = (const float*)d_in[16];
    const float* W1a = (const float*)d_in[17], *b1a = (const float*)d_in[18], *alpha1 = (const float*)d_in[19];
    const float* W2a = (const float*)d_in[20], *b2a = (const float*)d_in[21];
    const float* W1b = (const float*)d_in[22], *b1b = (const float*)d_in[23], *alpha2 = (const float*)d_in[24];
    const float* W2b = (const float*)d_in[25], *b2b = (const float*)d_in[26];

    float* out = (float*)d_out;                                // (B,T,C)
    float* SW  = out + (size_t)BATCH * SEQ * CH;               // (B,C,C)

    char* ws = (char*)d_ws;
    size_t off = 0;
    auto alloc = [&](size_t bytes) -> void* {
        void* p = ws + off;
        off += (bytes + 255) & ~(size_t)255;
        return p;
    };
    const size_t TC = (size_t)BATCH * SEQ * CH;                // 4M elements

    // bf16 weight copies ((out,in) row-major == GEMM (N,K) layout)
    unsigned short* wWq  = (unsigned short*)alloc((size_t)CH * CH * 2);
    unsigned short* wWk  = (unsigned short*)alloc((size_t)CH * CH * 2);
    unsigned short* wWv  = (unsigned short*)alloc((size_t)CH * CH * 2);
    unsigned short* wWo  = (unsigned short*)alloc((size_t)CH * CH * 2);
    unsigned short* wWqs = (unsigned short*)alloc((size_t)CH * SEQ * 2);
    unsigned short* wWks = (unsigned short*)alloc((size_t)CH * SEQ * 2);
    unsigned short* wW1a = (unsigned short*)alloc((size_t)FF * CH * 2);
    unsigned short* wW2a = (unsigned short*)alloc((size_t)CH * FF * 2);
    unsigned short* wW1b = (unsigned short*)alloc((size_t)FF * CH * 2);
    unsigned short* wW2b = (unsigned short*)alloc((size_t)CH * FF * 2);
    // activations
    unsigned short* bufLN  = (unsigned short*)alloc(TC * 2);   // ln_t1 / ln_t2 / ln_s / ln_f
    unsigned short* bufQ   = (unsigned short*)alloc(TC * 2);   // q   -> later qs
    unsigned short* bufK   = (unsigned short*)alloc(TC * 2);   // k   -> later ks
    unsigned short* bufV   = (unsigned short*)alloc(TC * 2);   // v
    unsigned short* bufO   = (unsigned short*)alloc(TC * 2);   // attn out -> later SW bf16
    float*          bufX   = (float*)alloc(TC * 4);            // x after out-proj -> later x2
    float*          bufTO  = (float*)alloc(TC * 4);            // temporal_out f32
    unsigned short* bufTOb = (unsigned short*)alloc(TC * 2);   // temporal_out bf16
    unsigned short* bufHid = (unsigned short*)alloc((size_t)BATCH * SEQ * FF * 2); // FFN hidden

    auto cvt = [&](const float* src, unsigned short* dst, long n) {
        k_cvt<<<1024, 256, 0, stream>>>(src, dst, n);
    };
    auto gemm = [&](const unsigned short* A, const unsigned short* Bw, const float* bias,
                    const float* ap, const float* res, float* oF, unsigned short* oB,
                    int M, int N, int K) {
        dim3 g(N / 128, M / 128);
        k_gemm<<<g, 256, 0, stream>>>(A, Bw, bias, ap, res, oF, oB, M, N, K);
    };

    // weight conversions
    cvt(Wq_t, wWq,  (long)CH * CH);
    cvt(Wk_t, wWk,  (long)CH * CH);
    cvt(Wv_t, wWv,  (long)CH * CH);
    cvt(Wo,   wWo,  (long)CH * CH);
    cvt(Wq_s, wWqs, (long)CH * SEQ);
    cvt(Wk_s, wWks, (long)CH * SEQ);
    cvt(W1a,  wW1a, (long)FF * CH);
    cvt(W2a,  wW2a, (long)CH * FF);
    cvt(W1b,  wW1b, (long)FF * CH);
    cvt(W2b,  wW2b, (long)CH * FF);

    const int MT = BATCH * SEQ;   // 4096 rows (temporal) ; spatial also B*C = 4096

    // ---- temporal branch ----
    k_layernorm<<<MT, 256, 0, stream>>>(x_T, ln_t1_w, ln_t1_b, bufLN, CH);
    gemm(bufLN, wWq, nullptr, nullptr, nullptr, nullptr, bufQ, MT, CH, CH);
    gemm(bufLN, wWk, nullptr, nullptr, nullptr, nullptr, bufK, MT, CH, CH);
    gemm(bufLN, wWv, nullptr, nullptr, nullptr, nullptr, bufV, MT, CH, CH);
    k_flash<<<BATCH * NH * (SEQ / 64), 128, 0, stream>>>(bufQ, bufK, bufV, bufO);
    gemm(bufO, wWo, nullptr, nullptr, x_T, bufX, nullptr, MT, CH, CH);      // x = x_T + o@Wo^T
    k_layernorm<<<MT, 256, 0, stream>>>(bufX, ln_t2_w, ln_t2_b, bufLN, CH);
    gemm(bufLN, wW1a, b1a, alpha1, nullptr, nullptr, bufHid, MT, FF, CH);   // PReLU hidden
    gemm(bufHid, wW2a, b2a, nullptr, bufX, bufTO, bufTOb, MT, CH, FF);      // temporal_out

    // ---- spatial branch ----
    k_layernorm<<<BATCH * CH, 256, 0, stream>>>(x_S, ln_s_w, ln_s_b, bufLN, SEQ);
    gemm(bufLN, wWqs, nullptr, nullptr, nullptr, nullptr, bufQ, BATCH * CH, CH, SEQ);  // qs
    gemm(bufLN, wWks, nullptr, nullptr, nullptr, nullptr, bufK, BATCH * CH, CH, SEQ);  // ks
    k_spatial<<<BATCH * (CH / 32), 128, 0, stream>>>(bufQ, bufK, SW);

    // ---- fusion: x2 = TO + TO @ SW^T  (per-batch GEMM, B operand = SW (n,m)) ----
    cvt(SW, bufO, (long)BATCH * CH * CH);
    for (int b = 0; b < BATCH; ++b)
        gemm(bufTOb + (size_t)b * SEQ * CH, bufO + (size_t)b * CH * CH, nullptr, nullptr,
             bufTO + (size_t)b * SEQ * CH, bufX + (size_t)b * SEQ * CH, nullptr, SEQ, CH, CH);

    // ---- final FFN + residual -> d_out ----
    k_layernorm<<<MT, 256, 0, stream>>>(bufX, ln_f_w, ln_f_b, bufLN, CH);
    gemm(bufLN, wW1b, b1b, alpha2, nullptr, nullptr, bufHid, MT, FF, CH);
    gemm(bufHid, wW2b, b2b, nullptr, bufX, out, nullptr, MT, CH, FF);
}
